// MSLSTMCell_1967095022176
// MI455X (gfx1250) — matrix operand order repaired and checked by reading, not compiled, not run on hardware
//
#include <hip/hip_runtime.h>
#include <cstdint>
#include <cstddef>

#define BATCH   32768
#define DIN     1024
#define HDIM    512

typedef __attribute__((ext_vector_type(16))) __bf16 v16bf;
typedef __attribute__((ext_vector_type(8)))  float  v8f;
typedef __attribute__((ext_vector_type(8)))  unsigned short u16x8;
typedef __attribute__((ext_vector_type(4)))  unsigned short u16x4;
typedef __attribute__((ext_vector_type(4)))  unsigned int   v4u;
typedef __attribute__((ext_vector_type(8)))  int            v8i;
typedef __attribute__((ext_vector_type(4)))  int            v4i;

union BFrag { v16bf v; u16x8 h[2]; };

// float4 element by (compile-time) index
#define F4E(v, j) ((j) == 0 ? (v).x : (j) == 1 ? (v).y : (j) == 2 ? (v).z : (v).w)

// hardware f32 -> bf16 (RNE) via v_cvt
__device__ __forceinline__ unsigned short f2bf(float f) {
  return __builtin_bit_cast(unsigned short, (__bf16)f);
}

__device__ __forceinline__ float sigm(float z) {
  return 1.0f / (1.0f + __expf(-z));
}

// tanh via hardware exp: 2/(1+exp(-2z)) - 1  (saturates correctly at +-inf)
__device__ __forceinline__ float tanh_fast(float z) {
  return 2.0f / (1.0f + __expf(-2.0f * z)) - 1.0f;
}

// ---------------------------------------------------------------------------
// TDM: issue a 2-D bf16 tile load (tile_dim0=32 elems x tile_dim1=128 rows)
// from a [32768 x 1024] bf16 tensor (row stride 1024 elems) into LDS.
// Descriptor packing per CDNA5 ISA ch.8 (D# groups 0/1; groups 2/3 unused=0).
// ---------------------------------------------------------------------------
__device__ __forceinline__ void tdm_load_tile_bf16(const unsigned short* gsrc,
                                                   unsigned lds_off) {
  const unsigned long long ga = (unsigned long long)(uintptr_t)gsrc;
  const v4u g0 = {
    1u,                                                  // count=1, user, no gather
    lds_off,                                             // lds_addr (bytes)
    (unsigned)ga,                                        // global_addr[31:0]
    (unsigned)((ga >> 32) & 0x01FFFFFFull) | 0x80000000u // addr[56:32] | type=2<<30
  };
  const v8i g1 = {
    (int)0x00010000,   // wg_mask=0, data_size=1 (2 bytes)
    (int)0x04000000,   // tensor_dim0=1024: low16 -> bits[63:48]
    (int)0x80000000,   // tensor_dim0 hi16=0 | tensor_dim1=32768 low16 -> bits[95:80]
    (int)0x00200000,   // tensor_dim1 hi16=0 | tile_dim0=32 -> bits[127:112]
    (int)128,          // tile_dim1=128; tile_dim2=0
    (int)1024,         // tensor_dim0_stride[31:0] = 1024 elems
    0, 0               // stride hi, tensor_dim1_stride=0 (2-D tile)
  };
  const v4i gz = { 0, 0, 0, 0 };
#if __clang_major__ >= 23
  const v8i gz8 = { 0, 0, 0, 0, 0, 0, 0, 0 };
  __builtin_amdgcn_tensor_load_to_lds(g0, g1, gz, gz, gz8, 0);
#else
  __builtin_amdgcn_tensor_load_to_lds(g0, g1, gz, gz, 0);
#endif
}

// ---------------------------------------------------------------------------
// Kernel 1: fused 5-gate GEMM + activations + cell update.
// Block = 320 threads (10 waves). Block tile: M=128 rows x 32 h-cols x 5 gates.
// Wave w: gate g=w/2, M-half mh=w&1 -> 64x32 patch = 4x2 WMMA 16x16 frags.
// Software-pipelined: next K-tile's global loads issue during current WMMAs.
// ---------------------------------------------------------------------------
__global__ __launch_bounds__(320)
void mslstm_gates(const float* __restrict__ x,   const float* __restrict__ hst,
                  const float* __restrict__ cin, const float* __restrict__ Cin,
                  const float* __restrict__ Wi,  const float* __restrict__ Wf,
                  const float* __restrict__ Wo,  const float* __restrict__ Wg1,
                  const float* __restrict__ Wg2,
                  const float* __restrict__ bi,  const float* __restrict__ bfv,
                  const float* __restrict__ bo,  const float* __restrict__ bg1,
                  const float* __restrict__ bg2,
                  float* __restrict__ out_o,   // d_out slot 0: o now, new_h after kernel2
                  float* __restrict__ out_c,   float* __restrict__ out_C,
                  unsigned short* __restrict__ cc)  // [B][1024] bf16 = [new_c | new_C]
{
  __shared__ __align__(16) unsigned char smem[81920];
  unsigned short* sA = (unsigned short*)smem;           // 128x32 bf16 (8 KB)
  unsigned short* sB = (unsigned short*)(smem + 8192);  // 5x2x32x16 bf16 (10 KB), frag-packed
  float*          zb = (float*)smem;                    // [5][128][32] f32 (80 KB), after K loop

  const int tid   = threadIdx.x;
  const int wave  = tid >> 5;
  const int lane  = tid & 31;
  const int g     = wave >> 1;          // gate 0..4
  const int mh    = wave & 1;           // M half
  const int cbase = blockIdx.x * 32;    // h-column base
  const int mbase = blockIdx.y * 128;   // row base

  v8f acc[4][2] = {};

  // ---- A staging: first 256 threads (8 waves) stage two 8-col chunks each
  const bool doA = (tid < 256);         // wave-uniform
  int arow[2], ach[2];
  #pragma unroll
  for (int i = 0; i < 2; ++i) {
    const int task = tid + i * 256;     // 512 chunk tasks (128 rows x 4 chunks)
    arow[i] = task >> 2;
    ach[i]  = (task & 3) << 3;
  }

  // ---- B staging: exactly one 4k x 4n task per thread (5*32*32/16 = 320)
  const int btg  = tid >> 6;                 // gate 0..4 (wave-uniform)
  const int bk0  = ((tid & 63) >> 3) << 2;   // 0,4,...,28
  const int bn0  = (tid & 7) << 2;           // 0,4,...,28
  const float* wsrc = (btg == 0) ? Wi : (btg == 1) ? Wf : (btg == 2) ? Wo
                    : (btg == 3) ? Wg1 : Wg2;
  const int bklane = bk0 & 16;
  const int bhidx  = bk0 & 15;
  const int btg2   = btg << 1;

  float4 ra[4];   // 2 chunks x 2 float4
  float4 rb[4];   // 4 k-rows x float4

  auto load_tiles = [&](int kk) {
    const float* abase = (kk < HDIM) ? x : hst;
    const int    ac    = (kk < HDIM) ? kk : (kk - HDIM);
    if (doA) {
      #pragma unroll
      for (int i = 0; i < 2; ++i) {
        const float* p = abase + (size_t)(mbase + arow[i]) * HDIM + ac + ach[i];
        ra[2 * i]     = *(const float4*)(p);
        ra[2 * i + 1] = *(const float4*)(p + 4);
      }
    }
    #pragma unroll
    for (int i = 0; i < 4; ++i)
      rb[i] = *(const float4*)(wsrc + (size_t)(kk + bk0 + i) * HDIM + cbase + bn0);
  };

  auto store_tiles = [&]() {
    if (doA) {
      #pragma unroll
      for (int i = 0; i < 2; ++i) {
        const float4 v0 = ra[2 * i], v1 = ra[2 * i + 1];
        u16x8 p = { f2bf(v0.x), f2bf(v0.y), f2bf(v0.z), f2bf(v0.w),
                    f2bf(v1.x), f2bf(v1.y), f2bf(v1.z), f2bf(v1.w) };
        *(u16x8*)(sA + arow[i] * 32 + ach[i]) = p;
      }
    }
    #pragma unroll
    for (int j = 0; j < 4; ++j) {        // 4 K-values per column -> one b64 store
      const int n = bn0 + j;
      u16x4 p = { f2bf(F4E(rb[0], j)), f2bf(F4E(rb[1], j)),
                  f2bf(F4E(rb[2], j)), f2bf(F4E(rb[3], j)) };
      *(u16x4*)(sB + ((btg2 + (n >> 4)) * 32 + (n & 15) + bklane) * 16 + bhidx) = p;
    }
  };

  load_tiles(0);

  for (int kk = 0; kk < DIN; kk += 32) {
    store_tiles();
    __syncthreads();

    if (kk + 32 < DIN) load_tiles(kk + 32);   // global loads overlap the WMMAs below

    BFrag bfrag[2];
    #pragma unroll
    for (int nt = 0; nt < 2; ++nt) {
      const unsigned short* bp = sB + (((g << 1) + nt) * 32 + lane) * 16;
      bfrag[nt].h[0] = *(const u16x8*)(bp);
      bfrag[nt].h[1] = *(const u16x8*)(bp + 8);
    }
    const int kb = (lane >> 4) << 4;   // A frag: lanes 0-15 K0-15, 16-31 K16-31
    #pragma unroll
    for (int mt = 0; mt < 4; ++mt) {
      const int m = mh * 64 + mt * 16 + (lane & 15);
      BFrag af;
      af.h[0] = *(const u16x8*)(sA + m * 32 + kb);
      af.h[1] = *(const u16x8*)(sA + m * 32 + kb + 8);
      #pragma unroll
      for (int nt = 0; nt < 2; ++nt) {
        acc[mt][nt] = __builtin_amdgcn_wmma_f32_16x16x32_bf16(
            false, af.v, false, bfrag[nt].v, (short)0, acc[mt][nt], false, false);
      }
    }
    __syncthreads();
  }

  // ---- stage pre-activations to LDS (arena reuse: sA/sB are dead)
  #pragma unroll
  for (int mt = 0; mt < 4; ++mt)
    #pragma unroll
    for (int nt = 0; nt < 2; ++nt)
      #pragma unroll
      for (int v = 0; v < 8; ++v) {
        const int row = mh * 64 + mt * 16 + v + ((lane >> 4) << 3);
        const int col = nt * 16 + (lane & 15);
        zb[((g * 128) + row) * 32 + col] = acc[mt][nt][v];
      }
  __syncthreads();

  // ---- fused epilogue: sigmoid/tanh gates + cell update + bf16 cc emit
  for (int idx = tid; idx < 128 * 32; idx += 320) {
    const int m   = idx >> 5;
    const int cl  = idx & 31;
    const size_t r = (size_t)(mbase + m);
    const int  col = cbase + cl;
    const float zi  = zb[(0 * 128 + m) * 32 + cl] + bi[col];
    const float zf  = zb[(1 * 128 + m) * 32 + cl] + bfv[col];
    const float zo  = zb[(2 * 128 + m) * 32 + cl] + bo[col];
    const float zg1 = zb[(3 * 128 + m) * 32 + cl] + bg1[col];
    const float zg2 = zb[(4 * 128 + m) * 32 + cl] + bg2[col];
    const float iv = sigm(zi), fv = sigm(zf), ov = sigm(zo);
    const float g1 = tanh_fast(zg1), g2 = tanh_fast(zg2);
    const size_t e = r * HDIM + col;
    const float nc = fv * cin[e] + iv * g1;
    const float nC = fv * Cin[e] + iv * g2;
    out_c[e] = nc;
    out_C[e] = nC;
    out_o[e] = ov;
    cc[r * (2 * HDIM) + col]        = f2bf(nc);
    cc[r * (2 * HDIM) + HDIM + col] = f2bf(nC);
  }
}

// ---------------------------------------------------------------------------
// Kernel 2: new_h = o * tanh(cc @ W_mlp + b_mlp)
// Block = 256 threads (8 waves), tile 128x128; wave = 32x64 = 2x4 frags.
// A tile (bf16 pass-through) is DMA'd by the Tensor Data Mover into a
// double-buffered LDS tile; B (fp32->bf16) keeps the register pipeline.
// ---------------------------------------------------------------------------
__global__ __launch_bounds__(256)
void mslstm_mlp(const unsigned short* __restrict__ cc,
                const float* __restrict__ Wm, const float* __restrict__ bm,
                float* __restrict__ outh)
{
  __shared__ __align__(16) unsigned short sA2[2 * 128 * 32]; // 16 KB, TDM double buffer
  __shared__ __align__(16) unsigned short sB[8 * 32 * 16];   // 8 KB, frag-packed

  const int tid   = threadIdx.x;
  const int wave  = tid >> 5;
  const int lane  = tid & 31;
  const int mq    = wave >> 1;           // 0..3
  const int nh    = wave & 1;            // 0..1
  const int nbase = blockIdx.x * 128;
  const int mbase = blockIdx.y * 128;

  v8f acc[2][4] = {};

  // B staging: exactly one 4k x 4n task per thread (32*128/16 = 256)
  const int bk0 = (tid >> 5) << 2;     // 0,4,...,28
  const int bn0 = (tid & 31) << 2;     // 0,4,...,124
  const int bklane = bk0 & 16;
  const int bhidx  = bk0 & 15;

  float4 rB[4];

  auto loadB = [&](int kk) {
    #pragma unroll
    for (int i = 0; i < 4; ++i)
      rB[i] = *(const float4*)(Wm + (size_t)(kk + bk0 + i) * HDIM + nbase + bn0);
  };
  auto storeB = [&]() {
    #pragma unroll
    for (int j = 0; j < 4; ++j) {
      const int n = bn0 + j;
      u16x4 p = { f2bf(F4E(rB[0], j)), f2bf(F4E(rB[1], j)),
                  f2bf(F4E(rB[2], j)), f2bf(F4E(rB[3], j)) };
      *(u16x4*)(sB + ((n >> 4) * 32 + (n & 15) + bklane) * 16 + bhidx) = p;
    }
  };

  // LDS byte offset of the TDM buffers (low 32 bits of the generic address)
  const unsigned ldsA0 = (unsigned)(unsigned long long)(uintptr_t)&sA2[0];
  const unsigned short* gA = cc + (size_t)mbase * (2 * HDIM);

  if (wave == 0)
    tdm_load_tile_bf16(gA, ldsA0);     // tile 0 -> buffer 0
  loadB(0);

  for (int kk = 0; kk < 2 * HDIM; kk += 32) {
    const int it = kk >> 5;
    const unsigned short* sAcur = sA2 + (it & 1) * (128 * 32);

    storeB();
    if (wave == 0) {
      if (kk + 32 < 2 * HDIM) {
        // DMA next tile into the other buffer (last read two phases ago),
        // then require the current tile to be resident (<=1 outstanding).
        tdm_load_tile_bf16(gA + kk + 32,
                           ldsA0 + (unsigned)(((it + 1) & 1) * (128 * 32 * 2)));
        __builtin_amdgcn_s_wait_tensorcnt(1);
      } else {
        __builtin_amdgcn_s_wait_tensorcnt(0);
      }
    }
    __syncthreads();

    if (kk + 32 < 2 * HDIM) loadB(kk + 32);   // weight loads overlap the WMMAs

    BFrag bfrag[4];
    #pragma unroll
    for (int nt = 0; nt < 4; ++nt) {
      const unsigned short* bp = sB + ((nh * 4 + nt) * 32 + lane) * 16;
      bfrag[nt].h[0] = *(const u16x8*)bp;
      bfrag[nt].h[1] = *(const u16x8*)(bp + 8);
    }
    const int kb = (lane >> 4) << 4;
    #pragma unroll
    for (int mt = 0; mt < 2; ++mt) {
      const int m = mq * 32 + mt * 16 + (lane & 15);
      BFrag af;
      af.h[0] = *(const u16x8*)(sAcur + m * 32 + kb);
      af.h[1] = *(const u16x8*)(sAcur + m * 32 + kb + 8);
      #pragma unroll
      for (int nt = 0; nt < 4; ++nt)
        acc[mt][nt] = __builtin_amdgcn_wmma_f32_16x16x32_bf16(
            false, af.v, false, bfrag[nt].v, (short)0, acc[mt][nt], false, false);
    }
    __syncthreads();
  }

  // epilogue: new_h = o * tanh(z + b);  o was stashed in outh by kernel 1
  #pragma unroll
  for (int mt = 0; mt < 2; ++mt)
    #pragma unroll
    for (int nt = 0; nt < 4; ++nt)
      #pragma unroll
      for (int v = 0; v < 8; ++v) {
        const size_t row = (size_t)mbase + mq * 32 + mt * 16 + v + ((lane >> 4) << 3);
        const int    col = nbase + nh * 64 + nt * 16 + (lane & 15);
        const float  z   = acc[mt][nt][v] + bm[col];
        const size_t e   = row * HDIM + col;
        outh[e] = outh[e] * tanh_fast(z);
      }
}

extern "C" void kernel_launch(void* const* d_in, const int* in_sizes, int n_in,
                              void* d_out, int out_size, void* d_ws, size_t ws_size,
                              hipStream_t stream) {
  (void)in_sizes; (void)n_in; (void)out_size; (void)ws_size;
  const float* x   = (const float*)d_in[0];
  const float* h   = (const float*)d_in[1];
  const float* c   = (const float*)d_in[2];
  const float* C   = (const float*)d_in[3];
  const float* Wi  = (const float*)d_in[4];
  const float* bi  = (const float*)d_in[5];
  const float* Wf  = (const float*)d_in[6];
  const float* bf_ = (const float*)d_in[7];
  const float* Wo  = (const float*)d_in[8];
  const float* bo  = (const float*)d_in[9];
  const float* Wg1 = (const float*)d_in[10];
  const float* bg1 = (const float*)d_in[11];
  const float* Wg2 = (const float*)d_in[12];
  const float* bg2 = (const float*)d_in[13];
  const float* Wm  = (const float*)d_in[14];
  const float* bm  = (const float*)d_in[15];

  float* out_h = (float*)d_out;                       // gets o, then new_h
  float* out_c = out_h + (size_t)BATCH * HDIM;
  float* out_C = out_c + (size_t)BATCH * HDIM;
  unsigned short* cc = (unsigned short*)d_ws;         // [B][1024] bf16, 64 MB

  dim3 grid1(HDIM / 32, BATCH / 128);   // 16 x 256
  dim3 blk1(320);
  mslstm_gates<<<grid1, blk1, 0, stream>>>(x, h, c, C, Wi, Wf, Wo, Wg1, Wg2,
                                           bi, bf_, bo, bg1, bg2,
                                           out_h, out_c, out_C, cc);

  dim3 grid2(HDIM / 128, BATCH / 128);  // 4 x 256
  dim3 blk2(256);
  mslstm_mlp<<<grid2, blk2, 0, stream>>>(cc, Wm, bm, out_h);
}
